// NormalGRUCell_38714835206789
// MI455X (gfx1250) — matrix operand order repaired and checked by reading, not compiled, not run on hardware
//
#include <hip/hip_runtime.h>

#define BB 64
#define TT 24
#define NN 325
#define FF 64
#define HH 64
#define BN (BB * NN)   // 20800 rows
#define G3 (3 * HH)    // 192 gate columns

typedef __attribute__((ext_vector_type(16))) __bf16 v16bf;
typedef __attribute__((ext_vector_type(2)))  __bf16 v2bf;
typedef __attribute__((ext_vector_type(16))) unsigned short us16;
typedef __attribute__((ext_vector_type(8)))  unsigned int u8x;
typedef __attribute__((ext_vector_type(8)))  float v8f;
typedef __attribute__((ext_vector_type(4)))  float f4;

__device__ __forceinline__ unsigned short f2bf(float f) {
    // software round-to-nearest-even fp32 -> bf16
    unsigned int u = __builtin_bit_cast(unsigned int, f);
    u += 0x7fffu + ((u >> 16) & 1u);
    return (unsigned short)(u >> 16);
}

// Two fp32 -> one packed bf16x2 dword.
// 1) HW packed builtin if declared; 2) fptrunc casts (lets the backend pick a
//    native v_cvt_*bf16* if gfx1250 has one).
__device__ __forceinline__ unsigned int pk_bf16(float lo, float hi) {
#if __has_builtin(__builtin_amdgcn_cvt_pk_bf16_f32)
    v2bf p = __builtin_amdgcn_cvt_pk_bf16_f32(lo, hi);
    return __builtin_bit_cast(unsigned int, p);
#else
    v2bf p;
    p[0] = (__bf16)lo;
    p[1] = (__bf16)hi;
    return __builtin_bit_cast(unsigned int, p);
#endif
}

// Hardware tanh (CDNA5 TRANS op) with software fallback.
__device__ __forceinline__ float htanh(float x) {
#if __has_builtin(__builtin_amdgcn_tanhf)
    return __builtin_amdgcn_tanhf(x);
#elif __has_builtin(__builtin_amdgcn_tanh_f32)
    return __builtin_amdgcn_tanh_f32(x);
#else
    float e = __expf(-2.0f * fabsf(x));
    float t = (1.0f - e) / (1.0f + e);
    return __builtin_copysignf(t, x);
#endif
}

__device__ __forceinline__ float sigm(float x) {
    // sigmoid(x) = 0.5*tanh(0.5x) + 0.5  (mul + TRANS + fma)
    return __builtin_fmaf(0.5f, htanh(0.5f * x), 0.5f);
}

// A-matrix (16x32 bf16) per-lane load from an fp32 row.
// Layout: lanes 0-15 (M=lane): K = k0..k0+15 with k0=0
//         lanes 16-31        : same with k0=16.  chunk adds 32 to K.
__device__ __forceinline__ v16bf load_a_bf16(const float* __restrict__ row, int lane, int chunk) {
    const int k0 = ((lane & 16) ? 16 : 0) + 32 * chunk;
    f4 x0 = *(const f4*)(row + k0);
    f4 x1 = *(const f4*)(row + k0 + 4);
    f4 x2 = *(const f4*)(row + k0 + 8);
    f4 x3 = *(const f4*)(row + k0 + 12);
    u8x a;
    a[0] = pk_bf16(x0[0], x0[1]);
    a[1] = pk_bf16(x0[2], x0[3]);
    a[2] = pk_bf16(x1[0], x1[1]);
    a[3] = pk_bf16(x1[2], x1[3]);
    a[4] = pk_bf16(x2[0], x2[1]);
    a[5] = pk_bf16(x2[2], x2[3]);
    a[6] = pk_bf16(x3[0], x3[1]);
    a[7] = pk_bf16(x3[2], x3[3]);
    return __builtin_bit_cast(v16bf, a);
}

// B-matrix (32x16 bf16) per-lane load from bf16 weights W[gate_row][k] (K contiguous).
// Layout used here (same k order as A): lanes 0-15 (N=lane): K=0..15 (2 per VGPR); lanes 16-31: K=16..31.
__device__ __forceinline__ v16bf load_b_bf16(const unsigned short* __restrict__ w,
                                             int coltile, int chunk, int lane) {
    const int nc = lane & 15;
    const int kb = ((lane & 16) ? 16 : 0) + 32 * chunk;
    us16 b = *(const us16*)(w + (coltile * 16 + nc) * FF + kb);
    return __builtin_bit_cast(v16bf, b);
}

#define WMMA_BF16(A, Bm, C) \
    __builtin_amdgcn_wmma_f32_16x16x32_bf16(false, (A), false, (Bm), (short)0, (C), false, false)

__global__ __launch_bounds__(128) void gru_weights_to_bf16(const float* __restrict__ wih,
                                                           const float* __restrict__ whh,
                                                           unsigned short* __restrict__ wbf) {
    int i = blockIdx.x * blockDim.x + threadIdx.x;
    if (i < G3 * FF) {
        wbf[i]           = f2bf(wih[i]);
        wbf[G3 * FF + i] = f2bf(whh[i]);
    }
}

// One GRU time step. Each wave owns 16 consecutive rows of the BN=20800 batch.
// h_prev row address = hpbase + (b*hp_bstride + n)*64  (h0: bstride=NN; outputs: bstride=TT*NN)
__global__ __launch_bounds__(128) void gru_step_kernel(
    const float* __restrict__ xbase,          // data + t*NN*FF
    const float* __restrict__ hpbase,         // h0 or outputs + (t-1)*NN*HH
    float* __restrict__ obase,                // outputs + t*NN*HH
    const unsigned short* __restrict__ wih,   // bf16 [192][64]
    const unsigned short* __restrict__ whh,   // bf16 [192][64]
    const float* __restrict__ bih,
    const float* __restrict__ bhh,
    int hp_bstride) {
    const int lane  = threadIdx.x & 31;
    const int wave  = threadIdx.x >> 5;
    const int rbase = (blockIdx.x * 4 + wave) * 16;

    // ---- A operands (this wave's 16 rows), fp32 -> bf16 ----
    const int ra = rbase + (lane & 15);
    const int ba = ra / NN;
    const int na = ra - ba * NN;
    const float* xrow = xbase + (size_t)(ba * (TT * NN) + na) * FF;
    const float* hrow = hpbase + (size_t)(ba * hp_bstride + na) * HH;
    v16bf ax0 = load_a_bf16(xrow, lane, 0);
    v16bf ax1 = load_a_bf16(xrow, lane, 1);
    v16bf ah0 = load_a_bf16(hrow, lane, 0);
    v16bf ah1 = load_a_bf16(hrow, lane, 1);

    // ---- accumulators: r,z gates share x+h; n gate keeps xn/hn separate ----
    v8f accRZ[8], accXN[4], accHN[4];
    v8f zero = {};
#pragma unroll
    for (int j = 0; j < 8; ++j) accRZ[j] = zero;
#pragma unroll
    for (int j = 0; j < 4; ++j) { accXN[j] = zero; accHN[j] = zero; }

#pragma unroll
    for (int j = 0; j < 12; ++j) {
        v16bf bx0 = load_b_bf16(wih, j, 0, lane);
        v16bf bx1 = load_b_bf16(wih, j, 1, lane);
        v16bf bh0 = load_b_bf16(whh, j, 0, lane);
        v16bf bh1 = load_b_bf16(whh, j, 1, lane);
        if (j < 8) {
            v8f c = accRZ[j];
            c = WMMA_BF16(ax0, bx0, c);
            c = WMMA_BF16(ax1, bx1, c);
            c = WMMA_BF16(ah0, bh0, c);
            c = WMMA_BF16(ah1, bh1, c);
            accRZ[j] = c;
        } else {
            v8f cx = accXN[j - 8];
            cx = WMMA_BF16(ax0, bx0, cx);
            cx = WMMA_BF16(ax1, bx1, cx);
            accXN[j - 8] = cx;
            v8f ch = accHN[j - 8];
            ch = WMMA_BF16(ah0, bh0, ch);
            ch = WMMA_BF16(ah1, bh1, ch);
            accHN[j - 8] = ch;
        }
    }

    // ---- elementwise GRU on the C-layout registers ----
    const int nl  = lane & 15;        // C/D: column within 16-wide tile
    const int hi8 = (lane >> 4) * 8;  // C/D: rows v (lanes 0-15) / v+8 (lanes 16-31)
    float brz[8], bxn[4], bhn[4];
#pragma unroll
    for (int j = 0; j < 8; ++j) brz[j] = bih[j * 16 + nl] + bhh[j * 16 + nl];
#pragma unroll
    for (int j = 0; j < 4; ++j) {
        bxn[j] = bih[128 + j * 16 + nl];
        bhn[j] = bhh[128 + j * 16 + nl];
    }

#pragma unroll
    for (int v = 0; v < 8; ++v) {
        const int rg = rbase + hi8 + v;
        const int bg = rg / NN;
        const int ng = rg - bg * NN;
        const float* hp = hpbase + (size_t)(bg * hp_bstride + ng) * HH;
        float* op = obase + (size_t)(bg * (TT * NN) + ng) * HH;
#pragma unroll
        for (int j = 0; j < 4; ++j) {
            const int col = j * 16 + nl;
            float r  = sigm(accRZ[j][v] + brz[j]);
            float z  = sigm(accRZ[j + 4][v] + brz[j + 4]);
            float xn = accXN[j][v] + bxn[j];
            float hn = accHN[j][v] + bhn[j];
            float cand  = htanh(__builtin_fmaf(r, hn, xn));
            float hprev = hp[col];
            // (1-z)*cand + z*hprev == cand + z*(hprev - cand)
            op[col] = __builtin_fmaf(z, hprev - cand, cand);
        }
    }
}

__global__ __launch_bounds__(256) void gru_copy_hlast(const float* __restrict__ out,
                                                      float* __restrict__ hlast) {
    int i = blockIdx.x * blockDim.x + threadIdx.x;
    if (i < BB * NN * HH) {
        int b = i / (NN * HH);
        int r = i - b * (NN * HH);
        hlast[i] = out[(size_t)(b * TT + (TT - 1)) * NN * HH + r];
    }
}

extern "C" void kernel_launch(void* const* d_in, const int* in_sizes, int n_in,
                              void* d_out, int out_size, void* d_ws, size_t ws_size,
                              hipStream_t stream) {
    (void)in_sizes; (void)n_in; (void)out_size; (void)ws_size;
    // setup_inputs order: 0 feature (unused by reference), 1 data, 2 h0,
    //                     3 w_ih, 4 w_hh, 5 b_ih, 6 b_hh
    const float* data = (const float*)d_in[1];
    const float* h0   = (const float*)d_in[2];
    const float* wih  = (const float*)d_in[3];
    const float* whh  = (const float*)d_in[4];
    const float* bih  = (const float*)d_in[5];
    const float* bhh  = (const float*)d_in[6];
    float* out = (float*)d_out;                       // [B,T,N,H] then h_last [B,N,H]
    unsigned short* wbf = (unsigned short*)d_ws;      // bf16 weights: 2 * 192*64 ushorts

    gru_weights_to_bf16<<<(G3 * FF + 127) / 128, 128, 0, stream>>>(wih, whh, wbf);

    for (int t = 0; t < TT; ++t) {
        const float* hp = (t == 0) ? h0 : (out + (size_t)(t - 1) * NN * HH);
        const int hbs   = (t == 0) ? NN : (TT * NN);
        gru_step_kernel<<<BN / 64, 128, 0, stream>>>(
            data + (size_t)t * NN * FF, hp, out + (size_t)t * NN * HH,
            wbf, wbf + G3 * FF, bih, bhh, hbs);
    }

    gru_copy_hlast<<<(BB * NN * HH + 255) / 256, 256, 0, stream>>>(
        out, out + (size_t)BB * TT * NN * HH);
}
